// RetinexMamba_61134564491467
// MI455X (gfx1250) — matrix operand order repaired and checked by reading, not compiled, not run on hardware
//
#include <hip/hip_runtime.h>
#include <hip/hip_fp16.h>

typedef __attribute__((ext_vector_type(16))) _Float16 v16h;
typedef __attribute__((ext_vector_type(8)))  _Float16 h8;
typedef __attribute__((ext_vector_type(8)))  float    v8f;
typedef __attribute__((ext_vector_type(4)))  unsigned int u32x4;
typedef __attribute__((ext_vector_type(8)))  int      i32x8;
typedef __attribute__((ext_vector_type(4)))  int      i32x4;

// ---------------------------------------------------------------------------
// activations: 0=none 1=gelu(exact erf) 2=silu 3=sigmoid 4=tanh
// ---------------------------------------------------------------------------
__device__ __forceinline__ float act_f(float v, int act) {
  switch (act) {
    case 1: return 0.5f * v * (1.0f + erff(v * 0.70710678118654752f));
    case 2: return v / (1.0f + expf(-v));
    case 3: return 1.0f / (1.0f + expf(-v));
    case 4: return tanhf(v);
    default: return v;
  }
}

// ---------------------------------------------------------------------------
// Generic strided GEMM:  C(m,n) = act( sum_k A(m,k)*B(k,n) + bias[n] ) + Res(m,n)
// A(m,k) = A[m*sam + k*sak]; B(k,n) = B[k*sbk + n*sbn]; C/Res likewise.
// WMMA f32<-f16, block tile 128x64, 8 waves (4 along M x 2 along N); each wave
// owns a 32x32 slab -> 2 A-frags x 2 B-frags = 4 v_wmma per K-step.
// LDS layouts are fragment-ready: A row-major [m][k], B transposed [n][k],
// so every lane's fragment is contiguous 16-B runs -> ds_load_b128.
// ---------------------------------------------------------------------------
#define BM 128
#define BN 64
#define BK 32
#define LDP (BK + 8)   // padded row: 40 f16 = 80 B (multiple of 16 B)

__device__ __forceinline__ v16h frag_a(const _Float16 As[][LDP], int row, int kh) {
  h8 lo = *(const h8*)&As[row][kh];
  h8 hi = *(const h8*)&As[row][kh + 16];
  return __builtin_shufflevector(lo, hi, 0,1,2,3,4,5,6,7,8,9,10,11,12,13,14,15);
}
__device__ __forceinline__ v16h frag_b(const _Float16 Bs[][LDP], int col, int kb) {
  h8 lo = *(const h8*)&Bs[col][kb];
  h8 hi = *(const h8*)&Bs[col][kb + 16];
  return __builtin_shufflevector(lo, hi, 0,1,2,3,4,5,6,7,8,9,10,11,12,13,14,15);
}

__global__ __launch_bounds__(256)
void k_gemm(const float* A, long sam, long sak,
            const float* B, long sbk, long sbn,
            const float* bias,
            const float* res, long srm, long srn,
            float* C, long scm, long scn,
            int M, int N, int K, int act)
{
  __shared__ __align__(16) _Float16 As[BM][LDP];   // As[m][k]
  __shared__ __align__(16) _Float16 Bs[BN][LDP];   // Bs[n][k]  (transposed)
  int tid  = threadIdx.x;
  int lane = tid & 31;
  int wid  = tid >> 5;
  int wm   = wid & 3;            // 4 waves along M, each covers 32 rows
  int wn   = wid >> 2;           // 2 waves along N, each covers 32 columns
  int m0 = blockIdx.y * BM, n0 = blockIdx.x * BN;

  v8f acc00 = {0.f,0.f,0.f,0.f,0.f,0.f,0.f,0.f};
  v8f acc01 = acc00, acc10 = acc00, acc11 = acc00;

  // staging map: thread -> (row, 8-wide k-chunk); A covers rows sr and sr+64
  int sr = tid >> 2;             // 0..63
  int sc = (tid & 3) * 8;        // 0,8,16,24
  bool fullM = (m0 + BM <= M);
  bool fullN = (n0 + BN <= N);

  for (int k0 = 0; k0 < K; k0 += BK) {
    bool fullK = (k0 + BK <= K);
    // ---- stage A tile (128x32): two 16-B stores per thread ----
    if (fullM && fullK) {
#pragma unroll
      for (int half = 0; half < 2; ++half) {
        int r = sr + half * 64;
        const float* ap = &A[(long)(m0 + r) * sam + (long)(k0 + sc) * sak];
        h8 hv;
#pragma unroll
        for (int j = 0; j < 8; ++j) hv[j] = (_Float16)ap[(long)j * sak];
        *(h8*)&As[r][sc] = hv;
      }
    } else {
#pragma unroll
      for (int half = 0; half < 2; ++half) {
        int r = sr + half * 64;
        h8 hv;
#pragma unroll
        for (int j = 0; j < 8; ++j) {
          int gm = m0 + r, gk = k0 + sc + j;
          hv[j] = (gm < M && gk < K)
                    ? (_Float16)A[(long)gm * sam + (long)gk * sak]
                    : (_Float16)0.f;
        }
        *(h8*)&As[r][sc] = hv;
      }
    }
    // ---- stage B tile (32x64) transposed into Bs[n][k] ----
    {
      h8 hv;
      if (fullN && fullK) {
        const float* bp = &B[(long)(k0 + sc) * sbk + (long)(n0 + sr) * sbn];
#pragma unroll
        for (int j = 0; j < 8; ++j) hv[j] = (_Float16)bp[(long)j * sbk];
      } else {
#pragma unroll
        for (int j = 0; j < 8; ++j) {
          int gk = k0 + sc + j, gn = n0 + sr;
          hv[j] = (gk < K && gn < N)
                    ? (_Float16)B[(long)gk * sbk + (long)gn * sbn]
                    : (_Float16)0.f;
        }
      }
      *(h8*)&Bs[sr][sc] = hv;
    }
    if (k0 + BK < K) {   // hint next tiles toward the caches (global_prefetch_b8)
      if (m0 + sr < M) __builtin_prefetch(&A[(long)(m0 + sr) * sam + (long)(k0 + BK) * sak], 0, 0);
      if (n0 + sr < N) __builtin_prefetch(&B[(long)(k0 + BK) * sbk + (long)(n0 + sr) * sbn], 0, 0);
    }
    __syncthreads();

    // ---- fragments: A lane row = (lane&15); lane<16 -> K runs [0..7],[16..23],
    // lane>=16 -> [8..15],[24..31].  B lane col = (lane&15), same K runs as A.
    int kh = (lane >> 4) << 3;
    int kb = (lane >> 4) << 3;
    int arow0 = wm * 32 + (lane & 15);
    int bc0   = wn * 32 + (lane & 15);
    v16h a0 = frag_a(As, arow0,      kh);
    v16h a1 = frag_a(As, arow0 + 16, kh);
    v16h b0 = frag_b(Bs, bc0,        kb);
    v16h b1 = frag_b(Bs, bc0 + 16,   kb);

    acc00 = __builtin_amdgcn_wmma_f32_16x16x32_f16(false, a0, false, b0,
                                                   (short)0, acc00, false, false);
    acc01 = __builtin_amdgcn_wmma_f32_16x16x32_f16(false, a0, false, b1,
                                                   (short)0, acc01, false, false);
    acc10 = __builtin_amdgcn_wmma_f32_16x16x32_f16(false, a1, false, b0,
                                                   (short)0, acc10, false, false);
    acc11 = __builtin_amdgcn_wmma_f32_16x16x32_f16(false, a1, false, b1,
                                                   (short)0, acc11, false, false);
    __syncthreads();
  }

  // C layout: lane n = lane&15 ; VGPR r -> M = r + 8*(lane>>4)
  int gn0 = n0 + wn * 32 + (lane & 15);
  int gn1 = gn0 + 16;
  int mb  = m0 + wm * 32 + ((lane >> 4) << 3);
#pragma unroll
  for (int mf = 0; mf < 2; ++mf) {
    v8f accA = mf ? acc10 : acc00;
    v8f accB = mf ? acc11 : acc01;
#pragma unroll
    for (int r2 = 0; r2 < 8; ++r2) {
      int gm = mb + mf * 16 + r2;
      if (gm < M) {
        if (gn0 < N) {
          float v = accA[r2];
          if (bias) v += bias[gn0];
          v = act_f(v, act);
          if (res) v += res[(long)gm * srm + (long)gn0 * srn];
          C[(long)gm * scm + (long)gn0 * scn] = v;
        }
        if (gn1 < N) {
          float v = accB[r2];
          if (bias) v += bias[gn1];
          v = act_f(v, act);
          if (res) v += res[(long)gm * srm + (long)gn1 * srn];
          C[(long)gm * scm + (long)gn1 * scn] = v;
        }
      }
    }
  }
}

// ---------------------------------------------------------------------------
// TDM / async-tensor compile probe (CDNA5 Tensor Data Mover).
// Present in the object to validate the tensor_load_to_lds + s_wait_tensorcnt
// + cluster-barrier ISA paths on gfx1250. NOT launched from the graph: the D#
// bitfield semantics cannot be verified without hardware, and a malformed
// descriptor must not be able to corrupt a timed run.
// ---------------------------------------------------------------------------
__global__ void k_tdm_probe(const float* gsrc, float* dummy)
{
  __shared__ float lds[1024];
  if (threadIdx.x == 0) {
    unsigned long long ga = (unsigned long long)(__SIZE_TYPE__)gsrc;
    u32x4 g0;
    g0[0] = 1u;                                          // count=1, user D#
    g0[1] = 0u;                                          // lds_addr = 0
    g0[2] = (unsigned)(ga & 0xffffffffu);                // global_addr[31:0]
    g0[3] = (unsigned)((ga >> 32) & 0x01ffffffu) | (2u << 30);  // addr hi, type=2
    i32x8 g1 = {0,0,0,0,0,0,0,0};
    g1[0] = (2 << 16);                                   // data_size = 4 B
    g1[2] = 1024;                                        // tensor_dim0
    g1[3] = (64 << 16) | 0;                              // tile_dim0 = 64
    i32x4 g2 = {0,0,0,0};
    i32x4 g3 = {0,0,0,0};
#if __clang_major__ >= 23
    i32x8 g4 = {0,0,0,0,0,0,0,0};
    __builtin_amdgcn_tensor_load_to_lds(g0, g1, g2, g3, g4, 0);
#else
    __builtin_amdgcn_tensor_load_to_lds(g0, g1, g2, g3, 0);
#endif
    __builtin_amdgcn_s_wait_tensorcnt(0);
  }
  __builtin_amdgcn_s_cluster_barrier();      // NOP when not in a cluster
  int cid = __builtin_amdgcn_cluster_id_x();
  __syncthreads();
  dummy[threadIdx.x] = lds[threadIdx.x] + (float)cid;
}

// ---------------------------------------------------------------------------
// im2col (generic conv) : col[m*K + k], k = (c,kh,kw)
// ---------------------------------------------------------------------------
__global__ void k_im2col(const float* src, float* col, int Cin, int H, int W,
                         int KH, int KW, int stride, int pad, int Ho, int Wo)
{
  long K = (long)Cin * KH * KW;
  long total = (long)Ho * Wo * K;
  long i = (long)blockIdx.x * blockDim.x + threadIdx.x;
  if (i >= total) return;
  long m = i / K; int k = (int)(i % K);
  int c = k / (KH * KW); int kr = k % (KH * KW);
  int kh = kr / KW, kw = kr % KW;
  int ho = (int)(m / Wo), wo = (int)(m % Wo);
  int h = ho * stride - pad + kh, w = wo * stride - pad + kw;
  col[i] = (h >= 0 && h < H && w >= 0 && w < W)
             ? src[((size_t)c * H + h) * W + w] : 0.f;
}

// im2col for ConvTranspose2d(k=4,s=2,p=1): out (2H,2W); contribution
// x[ih][iw]*w[ci][co][kh][kw] with ih=(ho+1-kh)/2 when even & in range.
__global__ void k_im2col_ct(const float* src, float* col, int Cin, int H, int W)
{
  int Ho = 2 * H, Wo = 2 * W;
  long K = (long)Cin * 16;
  long total = (long)Ho * Wo * K;
  long i = (long)blockIdx.x * blockDim.x + threadIdx.x;
  if (i >= total) return;
  long m = i / K; int k = (int)(i % K);
  int ci = k >> 4; int kk = k & 15;
  int kh = kk >> 2, kw = kk & 3;
  int ho = (int)(m / Wo), wo = (int)(m % Wo);
  int ph = ho + 1 - kh, pw = wo + 1 - kw;
  float v = 0.f;
  if (ph >= 0 && pw >= 0 && !(ph & 1) && !(pw & 1)) {
    int ih = ph >> 1, iw = pw >> 1;
    if (ih < H && iw < W) v = src[((size_t)ci * H + ih) * W + iw];
  }
  col[i] = v;
}

// repack ConvTranspose weight (Cin,Cout,4,4) -> wr[k*Cout + n], k=(ci,kh,kw)
__global__ void k_repack_ctw(const float* w, float* wr, int Cin, int Cout)
{
  long total = (long)Cin * 16 * Cout;
  long i = (long)blockIdx.x * blockDim.x + threadIdx.x;
  if (i >= total) return;
  long k = i / Cout; int n = (int)(i % Cout);
  int ci = (int)(k >> 4); int kk = (int)(k & 15);
  wr[i] = w[((size_t)ci * Cout + n) * 16 + kk];
}

// ---------------------------------------------------------------------------
// LayerNorm over channel dim for [C][M] planes
// ---------------------------------------------------------------------------
__global__ void k_ln2d(const float* x, const float* g, const float* b,
                       float* y, int C, int M)
{
  int m = blockIdx.x * blockDim.x + threadIdx.x;
  if (m >= M) return;
  float s = 0.f, s2 = 0.f;
  for (int c = 0; c < C; ++c) {
    float v = x[(size_t)c * M + m];
    s += v; s2 += v * v;
  }
  float mean = s / C;
  float var  = s2 / C - mean * mean;
  float inv  = rsqrtf(var + 1e-5f);
  for (int c = 0; c < C; ++c) {
    float v = x[(size_t)c * M + m];
    y[(size_t)c * M + m] = (v - mean) * inv * g[c] + b[c];
  }
}

// causal depthwise conv k=4 along rows of length L (+bias, SiLU)
__global__ void k_dwconv(const float* xi, const float* w, const float* bias,
                         float* out, int Din, int R, int L)
{
  long RL = (long)R * L;
  long total = Din * RL;
  long i = (long)blockIdx.x * blockDim.x + threadIdx.x;
  if (i >= total) return;
  int j = (int)(i / RL); long m = i % RL; int pos = (int)(m % L);
  float acc = bias[j];
#pragma unroll
  for (int t = 0; t < 4; ++t) {
    int p = pos - 3 + t;
    if (p >= 0) acc += w[j * 4 + t] * xi[(size_t)j * RL + m - 3 + t];
  }
  out[i] = acc / (1.0f + expf(-acc));      // silu
}

// delta[j][m] = softplus(d_in[m]*dt_w[j] + dt_b[j])
__global__ void k_delta(const float* din, const float* dtw, const float* dtb,
                        float* delta, int Din, long M)
{
  long total = Din * M;
  long i = (long)blockIdx.x * blockDim.x + threadIdx.x;
  if (i >= total) return;
  int j = (int)(i / M); long m = i % M;
  float v = din[m] * dtw[j] + dtb[j];
  delta[i] = (v > 20.f) ? v : log1pf(expf(v));
}

// selective scan: one block per row, one thread per Din channel, 16 states in regs
__global__ void k_scan(const float* delta, const float* xi,
                       const float* Bp, const float* Cp, const float* A_log,
                       float* ys, int Din, int L, long M)
{
  int row = blockIdx.x;
  int j = threadIdx.x;                 // blockDim.x == Din (>= 32)
  __shared__ float sB[16], sC[16];
  float A[16], h[16];
#pragma unroll
  for (int n = 0; n < 16; ++n) { A[n] = -expf(A_log[j * 16 + n]); h[n] = 0.f; }
  for (int pos = 0; pos < L; ++pos) {
    long m = (long)row * L + pos;
    __syncthreads();
    if (j < 16)        sB[j]      = Bp[(size_t)j * M + m];
    else if (j < 32)   sC[j - 16] = Cp[(size_t)(j - 16) * M + m];
    __syncthreads();
    float d  = delta[(size_t)j * M + m];
    float xv = xi[(size_t)j * M + m];
    float y = 0.f;
#pragma unroll
    for (int n = 0; n < 16; ++n) {
      float dA = expf(d * A[n]);
      h[n] = dA * h[n] + d * sB[n] * xv;
      y += h[n] * sC[n];
    }
    ys[(size_t)j * M + m] = y;
  }
}

// ys = (ys + xi*D[j]) * silu(z)
__global__ void k_gate(float* ys, const float* xi, const float* Dp,
                       const float* z, int Din, long M)
{
  long total = Din * M;
  long i = (long)blockIdx.x * blockDim.x + threadIdx.x;
  if (i >= total) return;
  int j = (int)(i / M);
  float zv = z[i];
  ys[i] = (ys[i] + xi[i] * Dp[j]) * (zv / (1.0f + expf(-zv)));
}

// per-channel plane transpose: src [C][Rr][Cc] -> dst[c][col*Rr + row]
__global__ void k_transpose(const float* src, float* dst, int C, int Rr, int Cc)
{
  long RC = (long)Rr * Cc;
  long total = C * RC;
  long i = (long)blockIdx.x * blockDim.x + threadIdx.x;
  if (i >= total) return;
  int c = (int)(i / RC); long rem = i % RC;
  int r = (int)(rem / Cc), col = (int)(rem % Cc);
  dst[(size_t)c * RC + (size_t)col * Rr + r] = src[i];
}

__global__ void k_add(float* a, const float* b, long n)
{
  long i = (long)blockIdx.x * blockDim.x + threadIdx.x;
  if (i < n) a[i] += b[i];
}

__global__ void k_affine(const float* x, float* y, float a, float b, long n)
{
  long i = (long)blockIdx.x * blockDim.x + threadIdx.x;
  if (i < n) y[i] = a * x[i] + b;
}

__global__ void k_gatemul(float* x, const float* g, long n)   // x = x*g + x
{
  long i = (long)blockIdx.x * blockDim.x + threadIdx.x;
  if (i < n) x[i] = x[i] * g[i] + x[i];
}

__global__ void k_scale_ch(float* x, const float* s, int M, long n)
{
  long i = (long)blockIdx.x * blockDim.x + threadIdx.x;
  if (i < n) x[i] *= s[i / M];
}

__global__ void k_avgpool(const float* src, float* dst, int H, int W, int f)
{
  int Ho = H / f, Wo = W / f;
  int i = blockIdx.x * blockDim.x + threadIdx.x;
  if (i >= Ho * Wo) return;
  int ho = i / Wo, wo = i % Wo;
  float s = 0.f;
  for (int a = 0; a < f; ++a)
    for (int b = 0; b < f; ++b)
      s += src[(ho * f + a) * W + wo * f + b];
  dst[i] = s / (f * f);
}

__global__ void k_se_reduce(const float* x, float* y, int M)
{
  int c = blockIdx.x;
  __shared__ float sm[256];
  float a = 0.f;
  for (int m = threadIdx.x; m < M; m += 256) a += x[(size_t)c * M + m];
  sm[threadIdx.x] = a;
  __syncthreads();
  for (int s = 128; s > 0; s >>= 1) {
    if (threadIdx.x < s) sm[threadIdx.x] += sm[threadIdx.x + s];
    __syncthreads();
  }
  if (threadIdx.x == 0) y[c] = sm[0] / M;
}

__global__ void k_se_fc(const float* y, const float* w1, const float* w2,
                        float* s, int C, int r)
{
  __shared__ float t[32];
  int tid = threadIdx.x;
  if (tid < r) {
    float a = 0.f;
    for (int c = 0; c < C; ++c) a += w1[tid * C + c] * y[c];
    t[tid] = 0.5f * a * (1.0f + erff(a * 0.70710678118654752f));  // gelu
  }
  __syncthreads();
  if (tid < C) {
    float a = 0.f;
    for (int k = 0; k < r; ++k) a += w2[tid * r + k] * t[k];
    s[tid] = 1.0f / (1.0f + expf(-a));
  }
}

// ===========================================================================
// Host side
// ===========================================================================
struct Arena {
  char* base; size_t off;
  float* alloc(size_t elems) {
    float* p = (float*)(base + off);
    off += ((elems * sizeof(float)) + 255) & ~(size_t)255;
    return p;
  }
};

struct Ctx { hipStream_t s; float* col; };

static inline dim3 gsz(long n) { return dim3((unsigned)((n + 255) / 256)); }

static void gemm(hipStream_t s,
                 const float* A, long sam, long sak,
                 const float* B, long sbk, long sbn,
                 const float* bias,
                 const float* res, long srm, long srn,
                 float* C, long scm, long scn,
                 int M, int N, int K, int act)
{
  dim3 g((N + BN - 1) / BN, (M + BM - 1) / BM);
  k_gemm<<<g, 256, 0, s>>>(A, sam, sak, B, sbk, sbn, bias,
                           res, srm, srn, C, scm, scn, M, N, K, act);
}

static void conv2d_gen(Ctx& cx, const float* src, int Cin, int H, int W,
                       int KH, int KW, int stride, int pad,
                       const float* w, const float* bias,
                       float* dst, int Cout, int act,
                       const float* res, long srm, long srn)
{
  int Ho = (H + 2 * pad - KH) / stride + 1;
  int Wo = (W + 2 * pad - KW) / stride + 1;
  long M = (long)Ho * Wo;
  long K = (long)Cin * KH * KW;
  k_im2col<<<gsz(M * K), 256, 0, cx.s>>>(src, cx.col, Cin, H, W, KH, KW,
                                         stride, pad, Ho, Wo);
  gemm(cx.s, cx.col, K, 1, w, 1, K, bias, res, srm, srn,
       dst, 1, M, (int)M, Cout, (int)K, act);
}

// ---- parameter tree (JAX pytree flatten order: dict keys sorted, lists in order)
struct Cursor { void* const* in; int i;
                const float* next() { return (const float*)in[i++]; } };

struct MambaP { const float *A_log,*D,*conv_b,*conv_w,*dt_b,*dt_w,
                            *in_proj,*out_proj,*x_proj; };
struct SmP    { MambaP mh, mv; const float *nb,*ng,*pb,*pw; };
struct BlockP { const float *ca_w1,*ca_w2,*f1b,*f1w,*f2b,*f2w,
                            *ia_cb,*ia_cw,*ia_gb,*ia_gw,*n1b,*n1g,*n2b,*n2g;
                SmP sm; };
struct DW     { const float *b, *w; };

static MambaP readM(Cursor& c) {
  MambaP m;
  m.A_log = c.next(); m.D = c.next(); m.conv_b = c.next(); m.conv_w = c.next();
  m.dt_b = c.next(); m.dt_w = c.next(); m.in_proj = c.next();
  m.out_proj = c.next(); m.x_proj = c.next();
  return m;
}
static BlockP readB(Cursor& c) {
  BlockP b;
  b.ca_w1 = c.next(); b.ca_w2 = c.next();
  b.f1b = c.next(); b.f1w = c.next(); b.f2b = c.next(); b.f2w = c.next();
  b.ia_cb = c.next(); b.ia_cw = c.next(); b.ia_gb = c.next(); b.ia_gw = c.next();
  b.n1b = c.next(); b.n1g = c.next(); b.n2b = c.next(); b.n2g = c.next();
  b.sm.mh = readM(c); b.sm.mv = readM(c);
  b.sm.nb = c.next(); b.sm.ng = c.next(); b.sm.pb = c.next(); b.sm.pw = c.next();
  return b;
}

// ---- one Mamba direction over [C][R*L] channel-major planes (m = r*L + pos)
static void mamba_dir(Ctx cx, Arena ar, const float* xn, int C, int R, int L,
                      const MambaP& mp, float* out)
{
  long M = (long)R * L;
  int Din = 2 * C;
  float* xz = ar.alloc((size_t)4 * C * M);
  gemm(cx.s, xn, 1, M, mp.in_proj, 1, C, nullptr, nullptr, 0, 0,
       xz, 1, M, (int)M, 4 * C, C, 0);
  float* xi = ar.alloc((size_t)Din * M);
  k_dwconv<<<gsz(Din * M), 256, 0, cx.s>>>(xz, mp.conv_w, mp.conv_b,
                                           xi, Din, R, L);
  float* xdbl = ar.alloc((size_t)33 * M);
  gemm(cx.s, xi, 1, M, mp.x_proj, 1, Din, nullptr, nullptr, 0, 0,
       xdbl, 1, M, (int)M, 33, Din, 0);
  float* delta = ar.alloc((size_t)Din * M);
  k_delta<<<gsz(Din * M), 256, 0, cx.s>>>(xdbl, mp.dt_w, mp.dt_b,
                                          delta, Din, M);
  float* ys = ar.alloc((size_t)Din * M);
  k_scan<<<R, Din, 0, cx.s>>>(delta, xi, xdbl + M, xdbl + 17 * M,
                              mp.A_log, ys, Din, L, M);
  k_gate<<<gsz(Din * M), 256, 0, cx.s>>>(ys, xi, mp.D,
                                         xz + (size_t)Din * M, Din, M);
  gemm(cx.s, ys, 1, M, mp.out_proj, 1, Din, nullptr, nullptr, 0, 0,
       out, 1, M, (int)M, C, Din, 0);
}

static void spatial_mamba(Ctx cx, Arena ar, const float* xin, int C, int H,
                          int W, const SmP& p, float* out)
{
  long M = (long)H * W;
  float* xn = ar.alloc((size_t)C * M);
  k_ln2d<<<gsz(M), 256, 0, cx.s>>>(xin, p.ng, p.nb, xn, C, (int)M);
  float* cat = ar.alloc((size_t)2 * C * M);
  mamba_dir(cx, ar, xn, C, H, W, p.mh, cat);                // horizontal
  float* xnT = ar.alloc((size_t)C * M);
  k_transpose<<<gsz(C * M), 256, 0, cx.s>>>(xn, xnT, C, H, W);
  float* yvT = ar.alloc((size_t)C * M);
  mamba_dir(cx, ar, xnT, C, W, H, p.mv, yvT);               // vertical
  k_transpose<<<gsz(C * M), 256, 0, cx.s>>>(yvT, cat + (size_t)C * M, C, W, H);
  gemm(cx.s, cat, 1, M, p.pw, 1, 2 * C, p.pb, xin, 1, M,
       out, 1, M, (int)M, C, 2 * C, 0);                     // + residual(xin)
}

static void rm_block(Ctx cx, Arena ar, float* x, const float* il,
                     const BlockP& p, int C, int H, int W)
{
  long M = (long)H * W;
  float* xn1 = ar.alloc((size_t)C * M);
  k_ln2d<<<gsz(M), 256, 0, cx.s>>>(x, p.n1g, p.n1b, xn1, C, (int)M);
  float* smo = ar.alloc((size_t)C * M);
  spatial_mamba(cx, ar, xn1, C, H, W, p.sm, smo);
  k_add<<<gsz(C * M), 256, 0, cx.s>>>(x, smo, C * M);
  // illumination-guided attention
  float* cat2 = ar.alloc((size_t)(C + 1) * M);
  hipMemcpyAsync(cat2, x, (size_t)C * M * 4, hipMemcpyDeviceToDevice, cx.s);
  hipMemcpyAsync(cat2 + (size_t)C * M, il, (size_t)M * 4,
                 hipMemcpyDeviceToDevice, cx.s);
  float* t1 = ar.alloc((size_t)C * M);
  conv2d_gen(cx, cat2, C + 1, H, W, 3, 3, 1, 1, p.ia_cw, p.ia_cb,
             t1, C, 0, nullptr, 0, 0);
  float* gate = ar.alloc((size_t)C * M);
  conv2d_gen(cx, t1, C, H, W, 3, 3, 1, 1, p.ia_gw, p.ia_gb,
             gate, C, 3, nullptr, 0, 0);
  k_gatemul<<<gsz(C * M), 256, 0, cx.s>>>(x, gate, C * M);
  // SE channel attention
  float* yc = ar.alloc(C);
  float* sc = ar.alloc(C);
  k_se_reduce<<<C, 256, 0, cx.s>>>(x, yc, (int)M);
  int rr = (C / 16 > 1) ? C / 16 : 1;
  int bdim = ((C + 31) / 32) * 32;
  k_se_fc<<<1, bdim, 0, cx.s>>>(yc, p.ca_w1, p.ca_w2, sc, C, rr);
  k_scale_ch<<<gsz(C * M), 256, 0, cx.s>>>(x, sc, (int)M, C * M);
  // FFN
  float* xn2 = ar.alloc((size_t)C * M);
  k_ln2d<<<gsz(M), 256, 0, cx.s>>>(x, p.n2g, p.n2b, xn2, C, (int)M);
  float* t4 = ar.alloc((size_t)4 * C * M);
  gemm(cx.s, xn2, 1, M, p.f1w, 1, C, p.f1b, nullptr, 0, 0,
       t4, 1, M, (int)M, 4 * C, C, 1);
  gemm(cx.s, t4, 1, M, p.f2w, 1, 4 * C, p.f2b, x, 1, M,
       x, 1, M, (int)M, C, 4 * C, 0);
}

static void conv_transpose(Ctx cx, const float* src, int Cin, int H, int W,
                           const float* w, const float* bias, float* ctw,
                           float* dst, int Cout)
{
  long K = (long)Cin * 16;
  k_repack_ctw<<<gsz(K * Cout), 256, 0, cx.s>>>(w, ctw, Cin, Cout);
  long M = (long)(2 * H) * (2 * W);
  k_im2col_ct<<<gsz(M * K), 256, 0, cx.s>>>(src, cx.col, Cin, H, W);
  gemm(cx.s, cx.col, K, 1, ctw, (long)Cout, 1, bias, nullptr, 0, 0,
       dst, 1, M, (int)M, Cout, (int)K, 0);
}

// ===========================================================================
extern "C" void kernel_launch(void* const* d_in, const int* in_sizes, int n_in,
                              void* d_out, int out_size, void* d_ws,
                              size_t ws_size, hipStream_t stream)
{
  (void)in_sizes; (void)n_in; (void)out_size; (void)ws_size;
  const float* xin = (const float*)d_in[0];

  Cursor cur{d_in, 1};
  // flatten order: bott, dec, down, enc, inp_b, inp_w, o1b, o1w, o2b, o2w,
  //                retinex{...}, up
  BlockP bott[6];                for (int i = 0; i < 6; ++i) bott[i] = readB(cur);
  BlockP dec0[4], dec1[4], dec2[2];
  for (int i = 0; i < 4; ++i) dec0[i] = readB(cur);   // C=192
  for (int i = 0; i < 4; ++i) dec1[i] = readB(cur);   // C=96
  for (int i = 0; i < 2; ++i) dec2[i] = readB(cur);   // C=48
  DW down[3]; for (int i = 0; i < 3; ++i) { down[i].b = cur.next(); down[i].w = cur.next(); }
  BlockP enc0[2], enc1[4], enc2[4];
  for (int i = 0; i < 2; ++i) enc0[i] = readB(cur);   // C=48
  for (int i = 0; i < 4; ++i) enc1[i] = readB(cur);   // C=96
  for (int i = 0; i < 4; ++i) enc2[i] = readB(cur);   // C=192
  const float* inp_b = cur.next(); const float* inp_w = cur.next();
  const float* o1b = cur.next();   const float* o1w = cur.next();
  const float* o2b = cur.next();   const float* o2w = cur.next();
  const float *e1b = cur.next(), *e1w = cur.next(), *e2b = cur.next(),
              *e2w = cur.next(), *i1b = cur.next(), *i1w = cur.next(),
              *i2b = cur.next(), *i2w = cur.next();
  cur.next(); cur.next(); cur.next(); cur.next();      // r1b,r1w,r2b,r2w unused
  DW up[3]; for (int i = 0; i < 3; ++i) { up[i].b = cur.next(); up[i].w = cur.next(); }

  // ---- workspace layout
  const long M0 = 128 * 128;
  size_t off = 0;
  auto carve = [&](size_t elems) {
    float* p = (float*)((char*)d_ws + off);
    off += ((elems * sizeof(float)) + 255) & ~(size_t)255;
    return p;
  };
  float* colbuf = carve((size_t)26 * 1024 * 1024);  // worst-case im2col
  float* skip0  = carve((size_t)48 * M0);
  float* skip1  = carve((size_t)96 * 4096);
  float* skip2  = carve((size_t)192 * 1024);
  float* featA  = carve((size_t)48 * M0);
  float* featB  = carve((size_t)48 * M0);
  float* illum  = carve(M0);
  float* il1    = carve(4096);
  float* il2    = carve(1024);
  float* il3    = carve(256);
  float* ctw    = carve((size_t)6144 * 192);
  Arena ar0{(char*)d_ws + off, 0};
  Ctx cx{stream, colbuf};

  // ---- retinex decomposition (illumination only; reflect is unused)
  {
    Arena ar = ar0;
    float* x01 = ar.alloc((size_t)3 * M0);
    k_affine<<<gsz(3 * M0), 256, 0, stream>>>(xin, x01, 0.5f, 0.5f, 3 * M0);
    float* f1 = ar.alloc((size_t)48 * M0);
    conv2d_gen(cx, x01, 3, 128, 128, 3, 3, 1, 1, e1w, e1b, f1, 48, 1, nullptr, 0, 0);
    float* f2 = ar.alloc((size_t)48 * M0);
    conv2d_gen(cx, f1, 48, 128, 128, 3, 3, 1, 1, e2w, e2b, f2, 48, 1, nullptr, 0, 0);
    float* t24 = ar.alloc((size_t)24 * M0);
    conv2d_gen(cx, f2, 48, 128, 128, 3, 3, 1, 1, i1w, i1b, t24, 24, 1, nullptr, 0, 0);
    conv2d_gen(cx, t24, 24, 128, 128, 3, 3, 1, 1, i2w, i2b, illum, 1, 3, nullptr, 0, 0);
    k_affine<<<gsz(M0), 256, 0, stream>>>(illum, illum, 1.f, 0.01f, M0);
  }
  k_avgpool<<<gsz(4096), 256, 0, stream>>>(illum, il1, 128, 128, 2);
  k_avgpool<<<gsz(1024), 256, 0, stream>>>(illum, il2, 128, 128, 4);
  k_avgpool<<<gsz(256), 256, 0, stream>>>(illum, il3, 128, 128, 8);

  // ---- stem
  conv2d_gen(cx, xin, 3, 128, 128, 3, 3, 1, 1, inp_w, inp_b, featA, 48, 0, nullptr, 0, 0);

  // ---- encoder
  for (int j = 0; j < 2; ++j) rm_block(cx, ar0, featA, illum, enc0[j], 48, 128, 128);
  hipMemcpyAsync(skip0, featA, (size_t)48 * M0 * 4, hipMemcpyDeviceToDevice, stream);
  conv2d_gen(cx, featA, 48, 128, 128, 4, 4, 2, 1, down[0].w, down[0].b, featB, 96, 0, nullptr, 0, 0);

  for (int j = 0; j < 4; ++j) rm_block(cx, ar0, featB, il1, enc1[j], 96, 64, 64);
  hipMemcpyAsync(skip1, featB, (size_t)96 * 4096 * 4, hipMemcpyDeviceToDevice, stream);
  conv2d_gen(cx, featB, 96, 64, 64, 4, 4, 2, 1, down[1].w, down[1].b, featA, 192, 0, nullptr, 0, 0);

  for (int j = 0; j < 4; ++j) rm_block(cx, ar0, featA, il2, enc2[j], 192, 32, 32);
  hipMemcpyAsync(skip2, featA, (size_t)192 * 1024 * 4, hipMemcpyDeviceToDevice, stream);
  conv2d_gen(cx, featA, 192, 32, 32, 4, 4, 2, 1, down[2].w, down[2].b, featB, 384, 0, nullptr, 0, 0);

  // ---- bottleneck
  for (int j = 0; j < 6; ++j) rm_block(cx, ar0, featB, il3, bott[j], 384, 16, 16);

  // ---- decoder
  conv_transpose(cx, featB, 384, 16, 16, up[0].w, up[0].b, ctw, featA, 192);
  k_add<<<gsz((long)192 * 1024), 256, 0, stream>>>(featA, skip2, (long)192 * 1024);
  for (int j = 0; j < 4; ++j) rm_block(cx, ar0, featA, il2, dec0[j], 192, 32, 32);

  conv_transpose(cx, featA, 192, 32, 32, up[1].w, up[1].b, ctw, featB, 96);
  k_add<<<gsz((long)96 * 4096), 256, 0, stream>>>(featB, skip1, (long)96 * 4096);
  for (int j = 0; j < 4; ++j) rm_block(cx, ar0, featB, il1, dec1[j], 96, 64, 64);

  conv_transpose(cx, featB, 96, 64, 64, up[2].w, up[2].b, ctw, featA, 48);
  k_add<<<gsz((long)48 * M0), 256, 0, stream>>>(featA, skip0, (long)48 * M0);
  for (int j = 0; j < 2; ++j) rm_block(cx, ar0, featA, illum, dec2[j], 48, 128, 128);

  // ---- output head
  {
    Arena ar = ar0;
    float* t = ar.alloc((size_t)48 * M0);
    conv2d_gen(cx, featA, 48, 128, 128, 3, 3, 1, 1, o1w, o1b, t, 48, 1, nullptr, 0, 0);
    conv2d_gen(cx, t, 48, 128, 128, 3, 3, 1, 1, o2w, o2b, (float*)d_out, 3, 4, nullptr, 0, 0);
  }
}
